// GRUActorCriticModel_4020089389730
// MI455X (gfx1250) — matrix operand order repaired and checked by reading, not compiled, not run on hardware
//
#include <hip/hip_runtime.h>
#include <hip/hip_bf16.h>

#define BATCH 65536
#define HID 64
#define RSTR 68    // GRU LDS row stride (floats)
#define PSTR 148   // pooled conv1 tile row stride
#define O2STR 132  // conv2 output tile row stride
#define SLAB (16 * PSTR + 16 * O2STR)  // 4480 floats per wave

typedef __attribute__((ext_vector_type(16))) _Float16 v16h;
typedef __attribute__((ext_vector_type(8)))  float    v8f;

__device__ __forceinline__ void ldsfence() {
    asm volatile("s_wait_dscnt 0" ::: "memory");
}
__device__ __forceinline__ void async_fence() {
    asm volatile("s_wait_asynccnt 0" ::: "memory");
}
__device__ __forceinline__ float sigm(float x) { return 1.0f / (1.0f + __expf(-x)); }
__device__ __forceinline__ float tanh_fast(float x) {
    x = fminf(fmaxf(x, -15.0f), 15.0f);
    float e = __expf(-2.0f * x);
    return (1.0f - e) / (1.0f + e);
}

// ---------------------------------------------------------------------------
// Generic B-operand swizzle: W [N, kchunks*32] f32 row-major -> f16 blocks.
// ---------------------------------------------------------------------------
__global__ void swz_kernel(const float* __restrict__ W, _Float16* __restrict__ dst,
                           int nblk512, int kchunks) {
    int i = blockIdx.x * 256 + threadIdx.x;
    if (i >= nblk512 * 512) return;
    int blk = i >> 9, r = i & 511;
    int lane = r >> 4, j = r & 15;
    int tile = blk / kchunks, kc = blk % kchunks;
    int n = tile * 16 + (lane & 15);
    int k = kc * 32 + ((lane < 16) ? 0 : 8) + ((j < 8) ? j : j + 8);
    dst[i] = (_Float16)W[n * (kchunks * 32) + k];
}

// conv1 weights [16,3,2,2] OIHW -> B block, k = (ky*2+kx)*3 + c, zero-pad K 12..31
__global__ void swz_c1_kernel(const float* __restrict__ w1, _Float16* __restrict__ dst) {
    int i = blockIdx.x * 256 + threadIdx.x;
    if (i >= 512) return;
    int lane = i >> 4, j = i & 15;
    int n = lane & 15;
    int k = ((lane < 16) ? 0 : 8) + ((j < 8) ? j : j + 8);
    float v = 0.0f;
    if (k < 12) {
        int c = k % 3, pix = k / 3;
        int ky = pix >> 1, kx = pix & 1;
        v = w1[n * 12 + c * 4 + ky * 2 + kx];
    }
    dst[i] = (_Float16)v;
}

// ---------------------------------------------------------------------------
// Async global <-> LDS row movers (16 rows x 64 f32; 128B contiguous per lane
// on both sides, so one address setup + 8 x b128 with shared imm offset).
// ---------------------------------------------------------------------------
__device__ __forceinline__ void load_rows_async(float* buf, const float* g, int rstride, int lane) {
    int r = lane >> 1, h = lane & 1;
    const float* gp = g + r * rstride + h * 32;
    unsigned la = (unsigned)(size_t)(buf + r * RSTR + h * 32);
    #pragma unroll
    for (int i = 0; i < 8; i++)
        asm volatile("global_load_async_to_lds_b128 %0, %1, off offset:%2"
                     :: "v"(la), "v"(gp), "n"(i * 16) : "memory");
}
__device__ __forceinline__ void store_rows_async(float* g, const float* buf, int rstride, int lane) {
    int r = lane >> 1, h = lane & 1;
    float* gp = g + r * rstride + h * 32;
    unsigned la = (unsigned)(size_t)(buf + r * RSTR + h * 32);
    #pragma unroll
    for (int i = 0; i < 8; i++)
        asm volatile("global_store_async_from_lds_b128 %0, %1, off offset:%2"
                     :: "v"(gp), "v"(la), "n"(i * 16) : "memory");
}

// A-operand 16x32 f16 from LDS f32 rows (row stride `stride`), K chunk base k0.
__device__ __forceinline__ v16h read_A(const float* buf, int k0, int lane, int stride) {
    const float* p = buf + (lane & 15) * stride + k0 + ((lane < 16) ? 0 : 8);
    float4 a = *(const float4*)(p);
    float4 bq = *(const float4*)(p + 4);
    float4 c = *(const float4*)(p + 16);
    float4 d = *(const float4*)(p + 20);
    v16h r;
    r[0] = (_Float16)a.x;  r[1] = (_Float16)a.y;  r[2] = (_Float16)a.z;  r[3] = (_Float16)a.w;
    r[4] = (_Float16)bq.x; r[5] = (_Float16)bq.y; r[6] = (_Float16)bq.z; r[7] = (_Float16)bq.w;
    r[8] = (_Float16)c.x;  r[9] = (_Float16)c.y;  r[10] = (_Float16)c.z; r[11] = (_Float16)c.w;
    r[12] = (_Float16)d.x; r[13] = (_Float16)d.y; r[14] = (_Float16)d.z; r[15] = (_Float16)d.w;
    return r;
}

__device__ __forceinline__ v8f wmma1(v16h a, const _Float16* blk, float bias, int lane) {
    v8f c;
    #pragma unroll
    for (int i = 0; i < 8; i++) c[i] = bias;
    v16h b = *(const v16h*)(blk + lane * 16);
    return __builtin_amdgcn_wmma_f32_16x16x32_f16(false, a, false, b, (short)0, c, false, false);
}
__device__ __forceinline__ v8f wmma2(v16h a0, v16h a1, const _Float16* blk, float bias, int lane) {
    v8f c;
    #pragma unroll
    for (int i = 0; i < 8; i++) c[i] = bias;
    v16h b0 = *(const v16h*)(blk + lane * 16);
    c = __builtin_amdgcn_wmma_f32_16x16x32_f16(false, a0, false, b0, (short)0, c, false, false);
    v16h b1 = *(const v16h*)(blk + 512 + lane * 16);
    c = __builtin_amdgcn_wmma_f32_16x16x32_f16(false, a1, false, b1, (short)0, c, false, false);
    return c;
}
__device__ __forceinline__ v8f wmma4(const v16h* a, const _Float16* blk, float bias, int lane) {
    v8f c;
    #pragma unroll
    for (int i = 0; i < 8; i++) c[i] = bias;
    #pragma unroll
    for (int kc = 0; kc < 4; kc++) {
        v16h b = *(const v16h*)(blk + kc * 512 + lane * 16);
        c = __builtin_amdgcn_wmma_f32_16x16x32_f16(false, a[kc], false, b, (short)0, c, false, false);
    }
    return c;
}

__device__ __forceinline__ void storeC_relu(float* buf, v8f c, int t, int lane) {
    int nlo = lane & 15, rbase = (lane < 16) ? 0 : 8;
    #pragma unroll
    for (int i = 0; i < 8; i++)
        buf[(rbase + i) * RSTR + t * 16 + nlo] = fmaxf(c[i], 0.0f);
}

// conv2 im2col A gather: K = f*4 + ky*2 + kx; kb multiple of 4 so groups of 4
// map to constant offsets {0,1,3,4} at f*9.
__device__ __forceinline__ v16h readA_c2(const float* prow, int cellbase, int kb) {
    const float* q = prow + cellbase;
    int fb = kb >> 2;
    v16h a;
    #pragma unroll
    for (int gidx = 0; gidx < 4; gidx++) {
        int f = fb + (gidx & 1) + (gidx >> 1) * 4;
        const float* r = q + f * 9;
        a[gidx * 4 + 0] = (_Float16)r[0];
        a[gidx * 4 + 1] = (_Float16)r[1];
        a[gidx * 4 + 2] = (_Float16)r[3];
        a[gidx * 4 + 3] = (_Float16)r[4];
    }
    return a;
}

// One GRU cell for a 16-row tile (inb/hb = LDS rows, stride RSTR).
// h' is written into hb and async-stored to outg (drained by next async_fence).
__device__ __forceinline__ void gru_layer(const float* inb, float* hb,
                                          const _Float16* wl,
                                          const float* bih, const float* bhh,
                                          float* outg, int lane)
{
    const _Float16* wih = wl;
    const _Float16* whh = wl + 12288;
    v16h ax0 = read_A(inb, 0, lane, RSTR), ax1 = read_A(inb, 32, lane, RSTR);
    v16h ah0 = read_A(hb, 0, lane, RSTR),  ah1 = read_A(hb, 32, lane, RSTR);
    int nlo = lane & 15, rbase = (lane < 16) ? 0 : 8;
    #pragma unroll 1
    for (int t = 0; t < 4; t++) {
        v8f gir = wmma2(ax0, ax1, wih + (t    ) * 1024, bih[       t * 16 + nlo], lane);
        v8f ghr = wmma2(ah0, ah1, whh + (t    ) * 1024, bhh[       t * 16 + nlo], lane);
        v8f giz = wmma2(ax0, ax1, wih + (4 + t) * 1024, bih[ 64 +  t * 16 + nlo], lane);
        v8f ghz = wmma2(ah0, ah1, whh + (4 + t) * 1024, bhh[ 64 +  t * 16 + nlo], lane);
        v8f gin = wmma2(ax0, ax1, wih + (8 + t) * 1024, bih[128 +  t * 16 + nlo], lane);
        v8f ghn = wmma2(ah0, ah1, whh + (8 + t) * 1024, bhh[128 +  t * 16 + nlo], lane);
        #pragma unroll
        for (int i = 0; i < 8; i++) {
            float* hp = hb + (rbase + i) * RSTR + t * 16 + nlo;
            float hv = *hp;
            float r = sigm(gir[i] + ghr[i]);
            float z = sigm(giz[i] + ghz[i]);
            float n = tanh_fast(gin[i] + r * ghn[i]);
            *hp = (1.0f - z) * n + z * hv;   // same-wave LDS: in-order ok
        }
    }
    ldsfence();                              // h' landed in LDS
    store_rows_async(outg, hb, 256, lane);   // async LDS -> HBM
}

struct FParams {
    const float* obs;
    const float* memory;
    const _Float16* wblk;
    const float* b1; const float* b2; const float* b3;
    const float* bih[4];
    const float* bhh[4];
    const float* actor_b1;  const float* actor_w2;  const float* actor_b2;
    const float* critic_b1; const float* critic_w2; const float* critic_b2;
    float* out;
};

// ws layout (halves)
#define OFF_AW1 98304
#define OFF_CW1 102400
#define OFF_C1  106496
#define OFF_C2  107008
#define OFF_C3  109056

// ---------------------------------------------------------------------------
// Fused conv + GRU + heads: one wave32 per 16-sample tile, 2 waves per block.
// ---------------------------------------------------------------------------
__global__ void __launch_bounds__(64) fused_kernel(FParams p)
{
    __shared__ __align__(16) float lds[2 * SLAB];
    int lane = threadIdx.x & 31;
    int wv   = threadIdx.x >> 5;
    int m0   = (blockIdx.x * 2 + wv) * 16;

    float* slab  = lds + wv * SLAB;
    float* ptile = slab;                 // 16 x PSTR  (conv1 pooled, rows=samples)
    float* o2t   = slab + 16 * PSTR;     // 16 x O2STR (conv2 out, k-ordered)
    float* bufA  = slab;                 // 16 x RSTR (aliases ptile after conv)
    float* bufB  = slab + 16 * RSTR;

    int s = lane & 15;
    int nlo = s, rbase = (lane < 16) ? 0 : 8;
    const float* o = p.obs + (size_t)(m0 + s) * 147;
    __builtin_prefetch(o, 0, 1);              // warm WGP$ for the im2col re-reads
    __builtin_prefetch(o + 73, 0, 1);
    const _Float16* c1blk = p.wblk + OFF_C1;
    const _Float16* c2blk = p.wblk + OFF_C2;
    const _Float16* c3blk = p.wblk + OFF_C3;

    // ---- conv1 (K=12 padded to 32) + relu + maxpool, via WMMA ----
    float b1v = p.b1[nlo];
    #pragma unroll
    for (int py = 0; py < 3; py++)
    #pragma unroll
    for (int px = 0; px < 3; px++) {
        v8f m;
        #pragma unroll
        for (int i = 0; i < 8; i++) m[i] = -3.4e38f;
        #pragma unroll
        for (int dy = 0; dy < 2; dy++)
        #pragma unroll
        for (int dx = 0; dx < 2; dx++) {
            int y = 2 * py + dy, x = 2 * px + dx;
            int base = (y * 7 + x) * 3;
            v16h a;
            #pragma unroll
            for (int j = 0; j < 16; j++) a[j] = (_Float16)0.0f;
            if (lane < 16) {          // K 0..7: row y (6 floats) + row y+1 (2)
                a[0] = (_Float16)o[base + 0]; a[1] = (_Float16)o[base + 1];
                a[2] = (_Float16)o[base + 2]; a[3] = (_Float16)o[base + 3];
                a[4] = (_Float16)o[base + 4]; a[5] = (_Float16)o[base + 5];
                a[6] = (_Float16)o[base + 21]; a[7] = (_Float16)o[base + 22];
            } else {                  // K 8..11: row y+1 floats 2..5
                a[0] = (_Float16)o[base + 23]; a[1] = (_Float16)o[base + 24];
                a[2] = (_Float16)o[base + 25]; a[3] = (_Float16)o[base + 26];
            }
            v8f c = wmma1(a, c1blk, b1v, lane);
            #pragma unroll
            for (int i = 0; i < 8; i++) m[i] = fmaxf(m[i], fmaxf(c[i], 0.0f));
        }
        #pragma unroll
        for (int i = 0; i < 8; i++)       // C layout: col = filter nlo, row = sample
            ptile[(rbase + i) * PSTR + nlo * 9 + py * 3 + px] = m[i];
    }
    ldsfence();

    // ---- conv2 (M=16,N=32,K=64 per spatial position) ----
    float b2v0 = p.b2[nlo], b2v1 = p.b2[16 + nlo];
    const float* prow = ptile + s * PSTR;
    int koff = (lane < 16) ? 0 : 8;
    #pragma unroll 1
    for (int pos = 0; pos < 4; pos++) {
        int cellbase = (pos >> 1) * 3 + (pos & 1);
        v16h a0 = readA_c2(prow, cellbase, 0 + koff);
        v16h a1 = readA_c2(prow, cellbase, 32 + koff);
        #pragma unroll
        for (int g0 = 0; g0 < 2; g0++) {
            v8f c = wmma2(a0, a1, c2blk + g0 * 1024, g0 ? b2v1 : b2v0, lane);
            #pragma unroll
            for (int i = 0; i < 8; i++)   // k-ordered: col = g*4 + pos
                o2t[(rbase + i) * O2STR + (g0 * 16 + nlo) * 4 + pos] = fmaxf(c[i], 0.0f);
        }
    }
    ldsfence();

    // ---- conv3 (M=16,N=64,K=128) -> x rows into bufA (ptile now dead) ----
    v16h e[4];
    e[0] = read_A(o2t, 0, lane, O2STR);
    e[1] = read_A(o2t, 32, lane, O2STR);
    e[2] = read_A(o2t, 64, lane, O2STR);
    e[3] = read_A(o2t, 96, lane, O2STR);
    #pragma unroll 1
    for (int n0 = 0; n0 < 4; n0++) {
        v8f c = wmma4(e, c3blk + n0 * 2048, p.b3[n0 * 16 + nlo], lane);
        storeC_relu(bufA, c, n0, lane);
    }
    ldsfence();

    // ---- GRUs + heads ----
    const float* mem_in = p.memory + (size_t)m0 * 256;
    float* out_mem  = p.out + (size_t)BATCH * 4 + (size_t)m0 * 256;
    float* out_logp = p.out + (size_t)m0 * 3;
    float* out_val  = p.out + (size_t)BATCH * 3 + m0;

    // each async_fence also drains the previous layer's async store before
    // its source buffer is overwritten; s_endpgm drains the last one.
    load_rows_async(bufB, mem_in, 256, lane);
    async_fence();
    gru_layer(bufA, bufB, p.wblk + 0 * 24576, p.bih[0], p.bhh[0], out_mem +   0, lane);
    load_rows_async(bufA, mem_in + 64, 256, lane);
    async_fence();
    gru_layer(bufB, bufA, p.wblk + 1 * 24576, p.bih[1], p.bhh[1], out_mem +  64, lane);
    load_rows_async(bufB, mem_in + 128, 256, lane);
    async_fence();
    gru_layer(bufA, bufB, p.wblk + 2 * 24576, p.bih[2], p.bhh[2], out_mem + 128, lane);
    load_rows_async(bufA, mem_in + 192, 256, lane);
    async_fence();
    gru_layer(bufB, bufA, p.wblk + 3 * 24576, p.bih[3], p.bhh[3], out_mem + 192, lane);
    // embedding rows now in bufA

    v16h ae0 = read_A(bufA, 0, lane, RSTR), ae1 = read_A(bufA, 32, lane, RSTR);
    const _Float16* aw1 = p.wblk + OFF_AW1;
    const _Float16* cw1 = p.wblk + OFF_CW1;

    #pragma unroll 1
    for (int t = 0; t < 4; t++)
        storeC_relu(bufB, wmma2(ae0, ae1, aw1 + t * 1024, p.actor_b1[t * 16 + nlo], lane), t, lane);
    ldsfence();
    if (lane < 16) {
        const float* row = bufB + lane * RSTR;
        float l0 = p.actor_b2[0], l1 = p.actor_b2[1], l2 = p.actor_b2[2];
        #pragma unroll
        for (int k = 0; k < 64; k++) {
            float v = row[k];
            l0 += p.actor_w2[k] * v;
            l1 += p.actor_w2[64 + k] * v;
            l2 += p.actor_w2[128 + k] * v;
        }
        float mx  = fmaxf(l0, fmaxf(l1, l2));
        float lse = logf(__expf(l0 - mx) + __expf(l1 - mx) + __expf(l2 - mx));
        float* op = out_logp + lane * 3;
        op[0] = l0 - mx - lse; op[1] = l1 - mx - lse; op[2] = l2 - mx - lse;
    }

    #pragma unroll 1
    for (int t = 0; t < 4; t++)
        storeC_relu(bufB, wmma2(ae0, ae1, cw1 + t * 1024, p.critic_b1[t * 16 + nlo], lane), t, lane);
    ldsfence();
    if (lane < 16) {
        const float* row = bufB + lane * RSTR;
        float v = p.critic_b2[0];
        #pragma unroll
        for (int k = 0; k < 64; k++) v += p.critic_w2[k] * row[k];
        out_val[lane] = v;
    }
}

// ---------------------------------------------------------------------------
extern "C" void kernel_launch(void* const* d_in, const int* in_sizes, int n_in,
                              void* d_out, int out_size, void* d_ws, size_t ws_size,
                              hipStream_t stream)
{
    (void)in_sizes; (void)n_in; (void)out_size; (void)ws_size;
    _Float16* wblk = (_Float16*)d_ws;

    // 1) weight swizzle: f16 B-operand blocks in ws (~235 KB)
    const float* gateW[8] = {
        (const float*)d_in[8],  (const float*)d_in[9],    // g1 L0 wih, whh
        (const float*)d_in[12], (const float*)d_in[13],   // g1 L1
        (const float*)d_in[16], (const float*)d_in[17],   // g2 L0
        (const float*)d_in[20], (const float*)d_in[21] }; // g2 L1
    for (int i = 0; i < 8; i++)
        swz_kernel<<<48, 256, 0, stream>>>(gateW[i], wblk + i * 12288, 24, 2);
    swz_kernel<<<16, 256, 0, stream>>>((const float*)d_in[24], wblk + OFF_AW1, 8, 2);  // actor_w1
    swz_kernel<<<16, 256, 0, stream>>>((const float*)d_in[28], wblk + OFF_CW1, 8, 2);  // critic_w1
    swz_c1_kernel<<<2, 256, 0, stream>>>((const float*)d_in[2], wblk + OFF_C1);        // conv_w1
    swz_kernel<<<8, 256, 0, stream>>>((const float*)d_in[4], wblk + OFF_C2, 4, 2);     // conv_w2 [32,64]
    swz_kernel<<<32, 256, 0, stream>>>((const float*)d_in[6], wblk + OFF_C3, 16, 4);   // conv_w3 [64,128]

    // 2) fused conv + GRU + heads
    FParams p;
    p.obs    = (const float*)d_in[0];
    p.memory = (const float*)d_in[1];
    p.wblk   = wblk;
    p.b1 = (const float*)d_in[3];
    p.b2 = (const float*)d_in[5];
    p.b3 = (const float*)d_in[7];
    p.bih[0] = (const float*)d_in[10]; p.bhh[0] = (const float*)d_in[11];
    p.bih[1] = (const float*)d_in[14]; p.bhh[1] = (const float*)d_in[15];
    p.bih[2] = (const float*)d_in[18]; p.bhh[2] = (const float*)d_in[19];
    p.bih[3] = (const float*)d_in[22]; p.bhh[3] = (const float*)d_in[23];
    p.actor_b1  = (const float*)d_in[25];
    p.actor_w2  = (const float*)d_in[26];
    p.actor_b2  = (const float*)d_in[27];
    p.critic_b1 = (const float*)d_in[29];
    p.critic_w2 = (const float*)d_in[30];
    p.critic_b2 = (const float*)d_in[31];
    p.out = (float*)d_out;
    fused_kernel<<<BATCH / 32, 64, 0, stream>>>(p);
}
